// CompressedKVAttention_11708080848911
// MI455X (gfx1250) — matrix operand order repaired and checked by reading, not compiled, not run on hardware
//
#include <hip/hip_runtime.h>
#include <stdint.h>

typedef _Float16 v16h __attribute__((ext_vector_type(16)));
typedef _Float16 v8h  __attribute__((ext_vector_type(8)));
typedef float    v8f  __attribute__((ext_vector_type(8)));

#define B_      4
#define H_      32
#define HKV_    8
#define NREP_   4
#define QLEN_   16
#define KVLEN_  8192
#define D_      128
#define SPLITS_ 8
#define TOKSPLIT_ (KVLEN_ / SPLITS_)   // 1024 tokens per split
#define TILE_   64                     // tokens per LDS staging tile
#define ROWS_   (B_ * H_ * QLEN_)      // 2048 output rows
#define PSTR_   (D_ + 2)               // per-row partial: 128 floats O + m + l

// -------------------------------------------------------------------------
// Kernel 1: split-K flash-decode partials. One block = (b, kv_head, split).
// 4 waves (wave32) = the 4 GQA query heads sharing this kv head.
// -------------------------------------------------------------------------
__global__ __launch_bounds__(128) void attn_partial_kernel(
    const float*  __restrict__ q,     // [B][H][QLEN][D] f32
    const int8_t* __restrict__ kq,    // [B][HKV][KV][D] i8
    const int8_t* __restrict__ vq,    // [B][HKV][KV][D] i8
    const float*  __restrict__ ksc,   // [B][HKV][KV]
    const float*  __restrict__ kzp,
    const float*  __restrict__ vsc,
    const float*  __restrict__ vzp,
    float*        __restrict__ part)  // [SPLITS][ROWS][PSTR]
{
    __shared__ _Float16 Kf[TILE_][D_];        // dequantized K tile, row-major   (16 KB)
    __shared__ _Float16 Vf[D_][TILE_];        // dequantized V tile, TRANSPOSED  (16 KB)
    __shared__ _Float16 Pst[4][16][TILE_];    // per-wave probability staging    ( 8 KB)

    const int split = blockIdx.x % SPLITS_;
    const int bg    = blockIdx.x / SPLITS_;
    const int b     = bg / HKV_;
    const int g     = bg % HKV_;

    const int tid  = threadIdx.x;
    const int lane = tid & 31;
    const int wv   = tid >> 5;        // wave = rep index within GQA group
    const int hl   = lane >> 4;       // lane half (0/1): WMMA layout selector
    const int ln   = lane & 15;

    const int hq = g * NREP_ + wv;    // query head served by this wave

    // ---- Q -> A-fragments (16x32 f16 A layout), fold softmax scale ----------
    const float qscale = 0.08838834764831845f;  // 1/sqrt(128)
    v16h afq[4];
    {
        const long qbase = (((long)b * H_ + hq) * QLEN_ + ln) * D_;
        #pragma unroll
        for (int c = 0; c < 4; ++c) {
            #pragma unroll
            for (int j = 0; j < 16; ++j) {
                const int dd = 32 * c + 16 * hl + j;
                afq[c][j] = (_Float16)(q[qbase + dd] * qscale);
            }
        }
    }

    v8f oacc[8];                      // 16x128 output accumulator, C layout
    #pragma unroll
    for (int i = 0; i < 8; ++i) oacc[i] = (v8f){};
    float mrow[8], lrow[8];           // stats for rows r + 8*hl
    #pragma unroll
    for (int r = 0; r < 8; ++r) { mrow[r] = -3.0e38f; lrow[r] = 0.0f; }

    const long kvhead = ((long)b * HKV_ + g) * KVLEN_;

    for (int t = 0; t < TOKSPLIT_ / TILE_; ++t) {
        const int tok0 = split * TOKSPLIT_ + t * TILE_;

        __syncthreads();   // previous tile fully consumed before overwrite

        // ---- cooperative int8 -> f16 dequant staging ------------------------
        {
            const int  tok = tid >> 1;            // 0..63
            const int  dh  = (tid & 1) * 64;      // half of the 128-d row
            const long sidx = kvhead + tok0 + tok;
            const long rb   = sidx * (long)D_ + dh;
            const float ks = ksc[sidx], kz = kzp[sidx];
            const float vs = vsc[sidx], vz = vzp[sidx];
            const int* kp = (const int*)(kq + rb);
            const int* vp = (const int*)(vq + rb);
            #pragma unroll
            for (int i4 = 0; i4 < 16; ++i4) {
                const int kw = kp[i4];
                const int vw = vp[i4];
                #pragma unroll
                for (int bb = 0; bb < 4; ++bb) {
                    const int d = dh + i4 * 4 + bb;
                    const float kvv = (float)(int8_t)(kw >> (8 * bb));
                    const float vvv = (float)(int8_t)(vw >> (8 * bb));
                    Kf[tok][d] = (_Float16)((kvv - kz) * ks);
                    Vf[d][tok] = (_Float16)((vvv - vz) * vs);
                }
            }
            // speculative prefetch of next tile (global_prefetch_b8)
            __builtin_prefetch((const char*)kp + TILE_ * D_, 0, 1);
            __builtin_prefetch((const char*)vp + TILE_ * D_, 0, 1);
        }
        __syncthreads();

        // ---- S = Q K^T over 4 subtiles of 16 tokens (4 WMMAs each, K=128) --
        v8f st[4];
        #pragma unroll
        for (int n = 0; n < 4; ++n) {
            v8f acc = (v8f){};
            #pragma unroll
            for (int c = 0; c < 4; ++c) {
                // B frag: col = token (ln), rows K = d-chunk, lane-half selects K 0-15/16-31
                const v16h bk = *(const v16h*)&Kf[16 * n + ln][32 * c + 16 * hl];
                acc = __builtin_amdgcn_wmma_f32_16x16x32_f16(
                          false, afq[c], false, bk, (short)0, acc, false, false);
            }
            st[n] = acc;
        }

        // ---- online softmax: row max / rescale / exp ------------------------
        float nm[8], al[8];
        #pragma unroll
        for (int r = 0; r < 8; ++r) {
            float mx = fmaxf(fmaxf(st[0][r], st[1][r]), fmaxf(st[2][r], st[3][r]));
            #pragma unroll
            for (int msk = 1; msk <= 8; msk <<= 1)
                mx = fmaxf(mx, __shfl_xor(mx, msk));
            nm[r]   = fmaxf(mrow[r], mx);
            al[r]   = __expf(mrow[r] - nm[r]);
            mrow[r] = nm[r];
        }

        #pragma unroll
        for (int r = 0; r < 8; ++r) {
            float rs = 0.0f;
            #pragma unroll
            for (int n = 0; n < 4; ++n) {
                const float p = __expf(st[n][r] - nm[r]);
                rs += p;
                Pst[wv][r + 8 * hl][16 * n + ln] = (_Float16)p;
            }
            #pragma unroll
            for (int msk = 1; msk <= 8; msk <<= 1)
                rs += __shfl_xor(rs, msk);
            lrow[r] = lrow[r] * al[r] + rs;
        }

        #pragma unroll
        for (int c = 0; c < 8; ++c)
            #pragma unroll
            for (int r = 0; r < 8; ++r)
                oacc[c][r] *= al[r];

        // in-wave LDS round-trip: make P stores visible before A-frag gathers
        asm volatile("s_wait_dscnt 0" ::: "memory");

        // ---- O += P V : gather P as A frags, V^T frags are contiguous ------
        v16h ap[2];
        #pragma unroll
        for (int kk = 0; kk < 2; ++kk) {
            const v8h lo = *(const v8h*)&Pst[wv][ln][32 * kk + 16 * hl];
            const v8h hi = *(const v8h*)&Pst[wv][ln][32 * kk + 16 * hl + 8];
            #pragma unroll
            for (int i = 0; i < 8; ++i) { ap[kk][i] = lo[i]; ap[kk][i + 8] = hi[i]; }
        }
        #pragma unroll
        for (int c2 = 0; c2 < 8; ++c2) {
            #pragma unroll
            for (int kk = 0; kk < 2; ++kk) {
                const v16h bv = *(const v16h*)&Vf[16 * c2 + ln][32 * kk + 16 * hl];
                oacc[c2] = __builtin_amdgcn_wmma_f32_16x16x32_f16(
                               false, ap[kk], false, bv, (short)0, oacc[c2], false, false);
            }
        }
    }

    // ---- spill partials: unnormalized O, plus (m, l) per row ----------------
    #pragma unroll
    for (int r = 0; r < 8; ++r) {
        const int  row = r + 8 * hl;
        const long rid = ((long)b * H_ + hq) * QLEN_ + row;
        float* pr = part + ((long)split * ROWS_ + rid) * PSTR_;
        #pragma unroll
        for (int c2 = 0; c2 < 8; ++c2)
            pr[16 * c2 + ln] = oacc[c2][r];
        if (ln == 0) { pr[D_] = mrow[r]; pr[D_ + 1] = lrow[r]; }
    }
}

// -------------------------------------------------------------------------
// Kernel 2: merge split partials with log-sum-exp rescaling.
// One block per output row, one thread per head-dim element.
// -------------------------------------------------------------------------
__global__ __launch_bounds__(128) void attn_combine_kernel(
    const float* __restrict__ part, float* __restrict__ out)
{
    const int rid = blockIdx.x;      // 0..ROWS_-1
    const int d   = threadIdx.x;     // 0..127

    float M = -3.0e38f;
    #pragma unroll
    for (int s = 0; s < SPLITS_; ++s)
        M = fmaxf(M, part[((long)s * ROWS_ + rid) * PSTR_ + D_]);

    float num = 0.0f, den = 0.0f;
    #pragma unroll
    for (int s = 0; s < SPLITS_; ++s) {
        const float* pr = part + ((long)s * ROWS_ + rid) * PSTR_;
        const float wgt = __expf(pr[D_] - M);
        den += wgt * pr[D_ + 1];
        num += wgt * pr[d];
    }
    out[(long)rid * D_ + d] = num / den;
}

// -------------------------------------------------------------------------
extern "C" void kernel_launch(void* const* d_in, const int* in_sizes, int n_in,
                              void* d_out, int out_size, void* d_ws, size_t ws_size,
                              hipStream_t stream) {
    const float*  q   = (const float*)d_in[0];
    const int8_t* kq  = (const int8_t*)d_in[1];
    const int8_t* vq  = (const int8_t*)d_in[2];
    const float*  ksc = (const float*)d_in[3];
    const float*  kzp = (const float*)d_in[4];
    const float*  vsc = (const float*)d_in[5];
    const float*  vzp = (const float*)d_in[6];
    float* part = (float*)d_ws;   // SPLITS_*ROWS_*PSTR_ floats ~ 8.5 MB
    float* out  = (float*)d_out;

    attn_partial_kernel<<<dim3(B_ * HKV_ * SPLITS_), dim3(128), 0, stream>>>(
        q, kq, vq, ksc, kzp, vsc, vzp, part);
    attn_combine_kernel<<<dim3(ROWS_), dim3(128), 0, stream>>>(part, out);
}
